// InferTree_50646254354828
// MI455X (gfx1250) — matrix operand order repaired and checked by reading, not compiled, not run on hardware
//
#include <hip/hip_runtime.h>
#include <math.h>

#define BATCH 16384
#define DIM   2048
#define NROWS 1089          // 33 (W1) + 32*33 (W2)
#define NPAD  1152          // 9 * 128, padded with zero rows
#define BM    128
#define BN    128
#define BK    32
#define LDA   40            // LDS row stride (bf16 elems) for x tiles: 32 + 8 pad
#define LDB   40            // LDS row stride for W tiles

typedef __attribute__((ext_vector_type(16))) __bf16 v16bf;
typedef __attribute__((ext_vector_type(8)))  float  v8f;

union Frag { v16bf v; uint4 q[2]; };

__device__ __forceinline__ unsigned short f32_to_bf16(float f) {
    unsigned u = __float_as_uint(f);
    u += 0x7FFFu + ((u >> 16) & 1u);        // round-to-nearest-even
    return (unsigned short)(u >> 16);
}
__device__ __forceinline__ float bf16_to_f32(unsigned short h) {
    return __uint_as_float(((unsigned)h) << 16);
}

// ---------------------------------------------------------------------------
// Kernel 1: build padded concatenated weight matrix as split bf16 (hi, lo),
// plus concatenated bias vector.
// ---------------------------------------------------------------------------
__global__ void __launch_bounds__(256) prep_weights(
        const float* __restrict__ W1, const float* __restrict__ b1,
        const float* __restrict__ W2, const float* __restrict__ b2,
        unsigned short* __restrict__ Whi, unsigned short* __restrict__ Wlo,
        float* __restrict__ bcat) {
    int idx = blockIdx.x * 256 + threadIdx.x;     // covers NPAD*DIM exactly
    int n = idx / DIM;
    int k = idx - n * DIM;
    float v;
    if (n < 33)          v = W1[n * DIM + k];
    else if (n < NROWS)  v = W2[(size_t)(n - 33) * DIM + k];
    else                 v = 0.0f;
    unsigned short hi = f32_to_bf16(v);
    float lo = v - bf16_to_f32(hi);
    Whi[idx] = hi;
    Wlo[idx] = f32_to_bf16(lo);
    if (idx < NPAD) {
        float bb;
        if (idx < 33)          bb = b1[idx];
        else if (idx < NROWS)  bb = b2[idx - 33];
        else                   bb = 0.0f;
        bcat[idx] = bb;
    }
}

// ---------------------------------------------------------------------------
// Kernel 2: C[B, NPAD] = x @ Wcat^T  via split-bf16 WMMA (hi*hi + hi*lo + lo*hi)
// 256 threads = 8 waves; block tile 128x128; wave tile 32x64 (2x4 WMMA tiles).
// 24 v_wmma_f32_16x16x32_bf16 per wave per K-step.
// ---------------------------------------------------------------------------
__global__ void __launch_bounds__(256) gemm_bf16x2(
        const float* __restrict__ X,
        const unsigned short* __restrict__ Whi,
        const unsigned short* __restrict__ Wlo,
        float* __restrict__ C) {
    __shared__ unsigned short sXhi[BM * LDA];
    __shared__ unsigned short sXlo[BM * LDA];
    __shared__ unsigned short sWhi[BN * LDB];
    __shared__ unsigned short sWlo[BN * LDB];

    const int tid   = threadIdx.x;
    const int lane  = tid & 31;
    const int wave  = tid >> 5;
    const int waveM = wave & 3;              // 4 waves along M (32 rows each)
    const int waveN = wave >> 2;             // 2 waves along N (64 cols each)
    const int rowBlk = blockIdx.y * BM;
    const int colBlk = blockIdx.x * BN;
    const int half  = lane >> 4;
    const int l16   = lane & 15;

    v8f acc[2][4];
    v8f zero = {};
    #pragma unroll
    for (int mt = 0; mt < 2; ++mt)
        #pragma unroll
        for (int nt = 0; nt < 4; ++nt)
            acc[mt][nt] = zero;

    for (int k0 = 0; k0 < DIM; k0 += BK) {
        // ---- stage X tile (fp32 -> split bf16) : 128 rows x 32 k ----
        #pragma unroll
        for (int i = 0; i < 4; ++i) {
            int seg = tid + 256 * i;          // 1024 float4 segments
            int r   = seg >> 3;               // row 0..127
            int c4  = (seg & 7) << 2;         // col 0,4,...,28
            float4 xv = *reinterpret_cast<const float4*>(
                X + (size_t)(rowBlk + r) * DIM + k0 + c4);
            unsigned short* dh = &sXhi[r * LDA + c4];
            unsigned short* dl = &sXlo[r * LDA + c4];
            float xs[4] = {xv.x, xv.y, xv.z, xv.w};
            #pragma unroll
            for (int c = 0; c < 4; ++c) {
                unsigned short hi = f32_to_bf16(xs[c]);
                dh[c] = hi;
                dl[c] = f32_to_bf16(xs[c] - bf16_to_f32(hi));
            }
        }
        // ---- stage W tiles (already bf16) : 128 rows x 32 k, hi and lo ----
        #pragma unroll
        for (int i = 0; i < 2; ++i) {
            int seg = tid + 256 * i;          // 512 uint4 segments per array
            int r   = seg >> 2;               // row 0..127
            int c8  = (seg & 3) << 3;         // 0,8,16,24
            size_t off = (size_t)(colBlk + r) * DIM + k0 + c8;
            uint4 h = *reinterpret_cast<const uint4*>(Whi + off);
            uint4 l = *reinterpret_cast<const uint4*>(Wlo + off);
            *reinterpret_cast<uint4*>(&sWhi[r * LDB + c8]) = h;
            *reinterpret_cast<uint4*>(&sWlo[r * LDB + c8]) = l;
        }
        __syncthreads();

        // ---- A fragments (ISA 16-bit A 16x32 layout) ----
        Frag ah[2], al[2];
        #pragma unroll
        for (int mt = 0; mt < 2; ++mt) {
            int m  = waveM * 32 + mt * 16 + l16;
            int kh = half * 8;
            ah[mt].q[0] = *reinterpret_cast<const uint4*>(&sXhi[m * LDA + kh]);
            ah[mt].q[1] = *reinterpret_cast<const uint4*>(&sXhi[m * LDA + kh + 16]);
            al[mt].q[0] = *reinterpret_cast<const uint4*>(&sXlo[m * LDA + kh]);
            al[mt].q[1] = *reinterpret_cast<const uint4*>(&sXlo[m * LDA + kh + 16]);
        }
        // ---- per N-tile: load one (hi,lo) B pair, issue 6 WMMAs ----
        #pragma unroll
        for (int nt = 0; nt < 4; ++nt) {
            int n  = waveN * 64 + nt * 16 + l16;
            int kb = half * 8;
            Frag bh, bl;
            const uint4* ph = reinterpret_cast<const uint4*>(&sWhi[n * LDB + kb]);
            bh.q[0] = ph[0]; bh.q[1] = ph[2];
            const uint4* pl = reinterpret_cast<const uint4*>(&sWlo[n * LDB + kb]);
            bl.q[0] = pl[0]; bl.q[1] = pl[2];
            #pragma unroll
            for (int mt = 0; mt < 2; ++mt) {
                acc[mt][nt] = __builtin_amdgcn_wmma_f32_16x16x32_bf16(
                    false, ah[mt].v, false, bh.v, (short)0, acc[mt][nt], false, false);
                acc[mt][nt] = __builtin_amdgcn_wmma_f32_16x16x32_bf16(
                    false, ah[mt].v, false, bl.v, (short)0, acc[mt][nt], false, false);
                acc[mt][nt] = __builtin_amdgcn_wmma_f32_16x16x32_bf16(
                    false, al[mt].v, false, bh.v, (short)0, acc[mt][nt], false, false);
            }
        }
        __syncthreads();
    }

    // ---- write back (C/D f32 layout: lane N=l16, VGPR v -> M = v + half*8) ----
    #pragma unroll
    for (int mt = 0; mt < 2; ++mt)
        #pragma unroll
        for (int nt = 0; nt < 4; ++nt) {
            int row = rowBlk + waveM * 32 + mt * 16 + half * 8;
            int col = colBlk + waveN * 64 + nt * 16 + l16;
            float* p = C + (size_t)row * NPAD + col;
            #pragma unroll
            for (int v = 0; v < 8; ++v)
                p[(size_t)v * NPAD] = acc[mt][nt][v];
        }
}

// ---------------------------------------------------------------------------
// Kernel 3: epilogue. One wave per sample; lane j = inner node j.
// ---------------------------------------------------------------------------
__global__ void zero_pen(float* pen) { *pen = 0.0f; }

__global__ void __launch_bounds__(256) finalize_kernel(
        const float* __restrict__ C, const float* __restrict__ bcat,
        const int* __restrict__ labels, float* __restrict__ out,
        float* __restrict__ pen) {
    const int lane = threadIdx.x & 31;
    const int wave = threadIdx.x >> 5;
    const int b = blockIdx.x * 8 + wave;
    const int lab = labels[b];
    const int g = lab >> 5;                 // label / F2
    const int w = lab & 31;                 // label % F2
    const float* row = C + (size_t)b * NPAD;

    // layer-2 node j = lane: streaming log-sum-exp over 33 logits
    const int j = lane;
    const int sub = (g == j) ? (w + 1) : 0;
    float mx = -INFINITY, s = 0.0f, pick = 0.0f;
    #pragma unroll
    for (int k = 0; k < 33; ++k) {
        float v = row[33 + j * 33 + k] + bcat[33 + j * 33 + k];
        if (k >= 1)
            out[(size_t)b * 1024 + j * 32 + (k - 1)] = v;   // raw leaf logits
        if (k == sub) pick = v;
        float nm = fmaxf(mx, v);
        s = s * __expf(mx - nm) + __expf(v - nm);
        mx = nm;
    }
    float ce2 = (mx + __logf(s)) - pick;
    #pragma unroll
    for (int off = 16; off >= 1; off >>= 1)
        ce2 += __shfl_down(ce2, off, 32);

    if (lane == 0) {
        float m1 = -INFINITY, s1 = 0.0f, p1 = 0.0f;
        #pragma unroll
        for (int k = 0; k < 33; ++k) {
            float v = row[k] + bcat[k];
            if (k == g + 1) p1 = v;
            float nm = fmaxf(m1, v);
            s1 = s1 * __expf(m1 - nm) + __expf(v - nm);
            m1 = nm;
        }
        float ce1 = (m1 + __logf(s1)) - p1;
        atomicAdd(pen, (ce1 + 0.5f * ce2) * (1.0f / (float)BATCH));
    }
}

// ---------------------------------------------------------------------------
extern "C" void kernel_launch(void* const* d_in, const int* in_sizes, int n_in,
                              void* d_out, int out_size, void* d_ws, size_t ws_size,
                              hipStream_t stream) {
    const float* x      = (const float*)d_in[0];
    const int*   labels = (const int*)d_in[1];
    const float* W1     = (const float*)d_in[2];
    const float* b1     = (const float*)d_in[3];
    const float* W2     = (const float*)d_in[4];
    const float* b2     = (const float*)d_in[5];
    float* out = (float*)d_out;
    float* pen = out + (size_t)BATCH * 1024;      // penalty scalar slot

    // workspace carve-up (~81 MB total)
    char* p = (char*)d_ws;
    unsigned short* Whi = (unsigned short*)p;  p += (size_t)NPAD * DIM * 2;
    unsigned short* Wlo = (unsigned short*)p;  p += (size_t)NPAD * DIM * 2;
    float* bcat = (float*)p;                   p += (size_t)NPAD * sizeof(float);
    float* C    = (float*)p;                   // BATCH * NPAD fp32 logits

    prep_weights<<<(NPAD * DIM) / 256, 256, 0, stream>>>(W1, b1, W2, b2, Whi, Wlo, bcat);

    dim3 grid(NPAD / BN, BATCH / BM);
    gemm_bf16x2<<<grid, 256, 0, stream>>>(x, Whi, Wlo, C);

    zero_pen<<<1, 1, 0, stream>>>(pen);
    finalize_kernel<<<BATCH / 8, 256, 0, stream>>>(C, bcat, labels, out, pen);
}
